// CGRNN_19473381720864
// MI455X (gfx1250) — compile-verified, matrix operand order checked by reading, not run on hardware
//
#include <hip/hip_runtime.h>
#include <math.h>

#define B_   32
#define DIN  32
#define DM   96
#define NN   500
#define TT   16
#define CIN  128           // DIN + DM
#define ZC   640           // (ORDER*SUP_LEN+1)*CIN
#define NB   16000         // NN*B_
#define QKVD 288
#define SC   125           // attention s-chunk
#define NCHUNK 4
#define NHEADS 2
#define DH   48

typedef __attribute__((ext_vector_type(16))) __bf16 v16bf;
typedef __attribute__((ext_vector_type(8)))  float  v8f;

union FragBF { unsigned u[8]; v16bf v; };

__device__ __forceinline__ unsigned short f2bf(float f) {
  unsigned u = __float_as_uint(f);
  u += 0x7fffu + ((u >> 16) & 1u);          // round-to-nearest-even
  return (unsigned short)(u >> 16);
}

__device__ __forceinline__ unsigned pack2bf(float a, float b) {
  return (unsigned)f2bf(a) | ((unsigned)f2bf(b) << 16);
}

struct GemmP {
  const float* A; const float* Bm; float* C; const float* bias;
  int M, N, K, lda, ldb, ldc;
  int zdiv, biasMode, act;                  // bias: 0 none,1 per-row,2 per-col; act: 0,1 relu,2 sig,3 tanh
  float scale;
  long long a1, a2, b1, b2, c1, c2;         // batch offsets: off = (z/zdiv)*o1 + (z%zdiv)*o2
};

#define LSTR 40                              // LDS row stride (u16): b64-aligned per row

// C[M,N] = act(scale * A[M,K] x B(K,N) + bias), bf16 WMMA with f32 accumulate.
// TRANSB=1: B[n*ldb + k]; TRANSB=0: B[k*ldb + n].
template <int TRANSB>
__global__ __launch_bounds__(128) void gemm_wmma_bf16(GemmP p) {
  __shared__ unsigned short sA[64 * LSTR];   // 64 rows x 32 k
  __shared__ unsigned short sB[64 * LSTR];   // 64 cols(n) x 32 k
  const int z  = blockIdx.z;
  const int i1 = z / p.zdiv, i2 = z % p.zdiv;
  const float* __restrict__ A  = p.A  + (long long)i1 * p.a1 + (long long)i2 * p.a2;
  const float* __restrict__ Bm = p.Bm + (long long)i1 * p.b1 + (long long)i2 * p.b2;
  float*       __restrict__ C  = p.C  + (long long)i1 * p.c1 + (long long)i2 * p.c2;
  const int bm = blockIdx.y * 64, bn = blockIdx.x * 64;
  const int tid  = threadIdx.x;
  const int w    = tid >> 5, lane = tid & 31;
  const int wm   = (w >> 1) * 32, wn = (w & 1) * 32;
  const int rr   = lane & 15, hi = lane >> 4;
  const bool fullMN = (bm + 64 <= p.M) && (bn + 64 <= p.N);
  v8f acc[2][2] = {};

  for (int k0 = 0; k0 < p.K; k0 += 32) {
    if (fullMN && (k0 + 32 <= p.K)) {
      // ---- fast staging: b128 global loads, packed LDS stores ----
      if (k0 + 32 < p.K) {                   // prefetch next K-chunk
        int r2 = tid >> 1;
        if ((tid & 1) == 0) {
          __builtin_prefetch(A + (long long)(bm + r2) * p.lda + (k0 + 32), 0, 1);
        } else {
          if (TRANSB) __builtin_prefetch(Bm + (long long)(bn + r2) * p.ldb + (k0 + 32), 0, 1);
          else        __builtin_prefetch(Bm + (long long)(k0 + 32 + (r2 & 31)) * p.ldb + bn, 0, 1);
        }
      }
      #pragma unroll
      for (int i = 0; i < 4; ++i) {
        int idx = i * 128 + tid;             // 0..511
        int row = idx >> 3;                  // 0..63
        int k   = (idx & 7) * 4;             // 0,4,...,28
        float4 a4 = *(const float4*)(A + (long long)(bm + row) * p.lda + (k0 + k));
        *(uint2*)&sA[row * LSTR + k] = make_uint2(pack2bf(a4.x, a4.y), pack2bf(a4.z, a4.w));
        if (TRANSB) {
          float4 b4 = *(const float4*)(Bm + (long long)(bn + row) * p.ldb + (k0 + k));
          *(uint2*)&sB[row * LSTR + k] = make_uint2(pack2bf(b4.x, b4.y), pack2bf(b4.z, b4.w));
        } else {
          int kk = idx >> 4;                 // 0..31
          int n4 = (idx & 15) * 4;           // 0..60
          float4 b4 = *(const float4*)(Bm + (long long)(k0 + kk) * p.ldb + (bn + n4));
          sB[(n4 + 0) * LSTR + kk] = f2bf(b4.x);
          sB[(n4 + 1) * LSTR + kk] = f2bf(b4.y);
          sB[(n4 + 2) * LSTR + kk] = f2bf(b4.z);
          sB[(n4 + 3) * LSTR + kk] = f2bf(b4.w);
        }
      }
    } else {
      // ---- edge staging: clamped unconditional loads + select (no exec branches) ----
      #pragma unroll
      for (int i = 0; i < 16; ++i) {
        int e = tid + i * 128;
        int r = e >> 5, k = e & 31;
        int gk = k0 + k;
        int gm = bm + r, gn = bn + r;
        int ck = min(gk, p.K - 1);
        float av = A[(long long)min(gm, p.M - 1) * p.lda + ck];
        if (gm >= p.M || gk >= p.K) av = 0.f;
        int cn = min(gn, p.N - 1);
        float bv = TRANSB ? Bm[(long long)cn * p.ldb + ck]
                          : Bm[(long long)ck * p.ldb + cn];
        if (gn >= p.N || gk >= p.K) bv = 0.f;
        sA[r * LSTR + k] = f2bf(av);
        sB[r * LSTR + k] = f2bf(bv);
      }
    }
    __syncthreads();
    // ---- gather fragments per ISA 16-bit A/B layout (wave32) ----
    FragBF fa[2], fb[2];
    #pragma unroll
    for (int mi = 0; mi < 2; ++mi) {
      const unsigned* pa = (const unsigned*)&sA[(wm + mi * 16 + rr) * LSTR];
      const unsigned* pb = (const unsigned*)&sB[(wn + mi * 16 + rr) * LSTR];
      #pragma unroll
      for (int j = 0; j < 4; ++j) {
        fa[mi].u[j]     = pa[(hi * 8 + 2 * j) >> 1];        // K = hi*8 + 2j, +1
        fa[mi].u[4 + j] = pa[(16 + hi * 8 + 2 * j) >> 1];   // K = 16 + hi*8 + 2j, +1
        fb[mi].u[j]     = pb[(hi * 8 + 2 * j) >> 1];
        fb[mi].u[4 + j] = pb[(16 + hi * 8 + 2 * j) >> 1];
      }
    }
    #pragma unroll
    for (int mi = 0; mi < 2; ++mi)
      #pragma unroll
      for (int ni = 0; ni < 2; ++ni)
        acc[mi][ni] = __builtin_amdgcn_wmma_f32_16x16x32_bf16(
            false, fa[mi].v, false, fb[ni].v, (short)0, acc[mi][ni], false, false);
    __syncthreads();
  }

  // ---- epilogue: f32 C layout (lane<16: M=r, lane>=16: M=8+r; N=lane&15) ----
  const int nn0 = lane & 15, mh = (lane >> 4) * 8;
  if (fullMN) {
    #pragma unroll
    for (int mi = 0; mi < 2; ++mi)
      #pragma unroll
      for (int ni = 0; ni < 2; ++ni)
        #pragma unroll
        for (int r = 0; r < 8; ++r) {
          int gm = bm + wm + mi * 16 + mh + r;
          int gn = bn + wn + ni * 16 + nn0;
          float v = acc[mi][ni][r] * p.scale;
          if (p.biasMode == 1) v += p.bias[gm];
          else if (p.biasMode == 2) v += p.bias[gn];
          if (p.act == 1) v = fmaxf(v, 0.f);
          else if (p.act == 2) v = 1.f / (1.f + __expf(-v));
          else if (p.act == 3) v = tanhf(v);
          C[(long long)gm * p.ldc + gn] = v;
        }
  } else {
    #pragma unroll
    for (int mi = 0; mi < 2; ++mi)
      #pragma unroll
      for (int ni = 0; ni < 2; ++ni)
        #pragma unroll
        for (int r = 0; r < 8; ++r) {
          int gm = bm + wm + mi * 16 + mh + r;
          int gn = bn + wn + ni * 16 + nn0;
          if (gm < p.M && gn < p.N) {
            float v = acc[mi][ni][r] * p.scale;
            if (p.biasMode == 1) v += p.bias[gm];
            else if (p.biasMode == 2) v += p.bias[gn];
            if (p.act == 1) v = fmaxf(v, 0.f);
            else if (p.act == 2) v = 1.f / (1.f + __expf(-v));
            else if (p.act == 3) v = tanhf(v);
            C[(long long)gm * p.ldc + gn] = v;
          }
        }
  }
}

// ---------------- small kernels ----------------

__global__ void k_coeffs(float* sac, float* somac) {
  if (threadIdx.x == 0 && blockIdx.x == 0) {
    float b0 = sqrtf(2e-5f), b1 = sqrtf(1e-5f);
    float ac = 1.f;
    for (int i = 0; i <= 100; ++i) {
      float beta = b0 + (b1 - b0) * (float)i * 0.01f;
      ac *= (1.f - beta);
      sac[i]   = sqrtf(ac);
      somac[i] = sqrtf(1.f - ac);
    }
  }
}

__global__ void k_zero(float* p, int n) {
  int i = blockIdx.x * 256 + threadIdx.x;
  if (i < n) p[i] = 0.f;
}

// z slot0 = concat(x_t, h) along channel dim; layout (B, ZC, N)
__global__ void k_build_xg(const float* __restrict__ x, const float* __restrict__ h,
                           float* __restrict__ z, int t) {
  int i = blockIdx.x * 256 + threadIdx.x;
  if (i >= B_ * CIN * NN) return;
  int n = i % NN;
  int c = (i / NN) % CIN;
  int b = i / (NN * CIN);
  float v;
  if (c < DIN) v = x[(((b * DIN + c) * NN + n) * TT) + t];
  else         v = h[(b * DM + (c - DIN)) * NN + n];
  z[((long long)b * ZC + c) * NN + n] = v;
}

// h = u*h + (1-u)*c  (u,c pre-activated in GEMM epilogue); fused length-select
__global__ void k_combine(const float* __restrict__ u, const float* __restrict__ c,
                          float* __restrict__ h, float* __restrict__ out,
                          const int* __restrict__ lengths, int t) {
  int i = blockIdx.x * 256 + threadIdx.x;
  if (i >= B_ * DM * NN) return;
  int b = i / (DM * NN);
  float uu = u[i];
  float hn = uu * h[i] + (1.f - uu) * c[i];
  h[i] = hn;
  bool sel = (lengths[b] - 1 == t);
  if (t == 0) out[i] = sel ? hn : 0.f;
  else if (sel) out[i] = hn;
}

// inp[(n*B+b)*96+j] = h[b,j,n] + emb[n,j]*mask[b,0,n,t-1]
__global__ void k_build_inp(const float* __restrict__ h, const float* __restrict__ emb,
                            const float* __restrict__ mask, float* __restrict__ inp, int t) {
  int i = blockIdx.x * 256 + threadIdx.x;
  if (i >= NB * DM) return;
  int j = i % DM;
  int b = (i / DM) % B_;
  int n = i / (DM * B_);
  float m = mask[(b * NN + n) * TT + (t - 1)];
  inp[i] = h[(b * DM + j) * NN + n] + emb[n * DM + j] * m;
}

__global__ __launch_bounds__(128) void k_softmax(float* __restrict__ s, int rows, int cols) {
  int row  = blockIdx.x * 4 + (threadIdx.x >> 5);
  int lane = threadIdx.x & 31;
  if (row >= rows) return;
  float* p = s + (long long)row * cols;
  float m = -3.4e38f;
  for (int i = lane; i < cols; i += 32) m = fmaxf(m, p[i]);
  #pragma unroll
  for (int d = 16; d > 0; d >>= 1) m = fmaxf(m, __shfl_xor(m, d, 32));
  float sum = 0.f;
  for (int i = lane; i < cols; i += 32) { float e = __expf(p[i] - m); p[i] = e; sum += e; }
  #pragma unroll
  for (int d = 16; d > 0; d >>= 1) sum += __shfl_xor(sum, d, 32);
  float inv = 1.f / sum;
  for (int i = lane; i < cols; i += 32) p[i] *= inv;
}

// out = LN(a + r) over rows of 96; one wave32 per row (3 elems/lane)
__global__ __launch_bounds__(128) void k_add_ln(const float* __restrict__ a,
                                                const float* __restrict__ r,
                                                const float* __restrict__ g,
                                                const float* __restrict__ be,
                                                float* __restrict__ out, int rows) {
  int row  = blockIdx.x * 4 + (threadIdx.x >> 5);
  int lane = threadIdx.x & 31;
  if (row >= rows) return;
  const float* pa = a + (long long)row * DM;
  const float* pr = r + (long long)row * DM;
  float x0 = pa[lane]      + pr[lane];
  float x1 = pa[lane + 32] + pr[lane + 32];
  float x2 = pa[lane + 64] + pr[lane + 64];
  float s  = x0 + x1 + x2;
  float sq = x0 * x0 + x1 * x1 + x2 * x2;
  #pragma unroll
  for (int d = 16; d > 0; d >>= 1) { s += __shfl_xor(s, d, 32); sq += __shfl_xor(sq, d, 32); }
  float mean = s * (1.f / 96.f);
  float var  = sq * (1.f / 96.f) - mean * mean;
  float rs   = rsqrtf(var + 1e-5f);
  float* po = out + (long long)row * DM;
  po[lane]      = (x0 - mean) * rs * g[lane]      + be[lane];
  po[lane + 32] = (x1 - mean) * rs * g[lane + 32] + be[lane + 32];
  po[lane + 64] = (x2 - mean) * rs * g[lane + 64] + be[lane + 64];
}

__device__ __forceinline__ unsigned pcg(unsigned v) {
  v = v * 747796405u + 2891336453u;
  unsigned w = ((v >> ((v >> 28u) + 4u)) ^ v) * 277803737u;
  return (w >> 22u) ^ w;
}

// h = sac[ds]*h + somac[ds]*(mu + randn*sig); mu/sig laid out (N,B,96)
__global__ void k_noise(float* __restrict__ h, const float* __restrict__ mu,
                        const float* __restrict__ sig, const float* __restrict__ delta,
                        const float* __restrict__ sac, const float* __restrict__ somac, int t) {
  int i = blockIdx.x * 256 + threadIdx.x;
  if (i >= B_ * DM * NN) return;
  int n = i % NN;
  int j = (i / NN) % DM;
  int b = i / (NN * DM);
  int ds = (int)floorf(delta[b * TT + t] * 100.f);
  ds = min(max(ds, 0), 100);
  unsigned s0 = pcg((unsigned)i * 0x9E3779B9u + (unsigned)t * 0x85EBCA6Bu + 0x1234567u);
  unsigned s1 = pcg(s0 ^ 0xC2B2AE35u);
  float u1 = ((float)(s0 >> 8) + 1.f) * (1.f / 16777216.f);   // (0,1]
  float u2 = (float)(s1 >> 8) * (1.f / 16777216.f);
  float rn = sqrtf(-2.f * logf(u1)) * __cosf(6.28318530718f * u2);
  int nb = (n * B_ + b) * DM + j;
  h[i] = sac[ds] * h[i] + somac[ds] * (mu[nb] + rn * sig[nb]);
}

// ---------------- host side ----------------

static inline void launch_gemm(hipStream_t st, const float* A, const float* Bm, float* C,
    const float* bias, int M, int N, int K, int lda, int ldb, int ldc, int transB,
    int nz, int zdiv, long long a1, long long a2, long long b1, long long b2,
    long long c1, long long c2, int biasMode, int act, float scale) {
  GemmP p{A, Bm, C, bias, M, N, K, lda, ldb, ldc, zdiv, biasMode, act, scale,
          a1, a2, b1, b2, c1, c2};
  dim3 grid((unsigned)((N + 63) / 64), (unsigned)((M + 63) / 64), (unsigned)nz);
  if (transB) gemm_wmma_bf16<1><<<grid, dim3(128), 0, st>>>(p);
  else        gemm_wmma_bf16<0><<<grid, dim3(128), 0, st>>>(p);
}

extern "C" void kernel_launch(void* const* d_in, const int* in_sizes, int n_in,
                              void* d_out, int out_size, void* d_ws, size_t ws_size,
                              hipStream_t stream) {
  const float* x       = (const float*)d_in[0];
  const float* adj     = (const float*)d_in[1];
  const float* delta   = (const float*)d_in[2];
  const int*   lengths = (const int*)  d_in[3];
  /* d_in[4] time_encoding: unused by reference */
  const float* mask    = (const float*)d_in[5];
  const float* emb     = (const float*)d_in[6];
  const float* Wqkv    = (const float*)d_in[7];
  const float* bqkv    = (const float*)d_in[8];
  const float* Wo      = (const float*)d_in[9];
  const float* bo      = (const float*)d_in[10];
  const float* ln1g    = (const float*)d_in[11];
  const float* ln1b    = (const float*)d_in[12];
  const float* ln2g    = (const float*)d_in[13];
  const float* ln2b    = (const float*)d_in[14];
  const float* W1      = (const float*)d_in[15];
  const float* b1v     = (const float*)d_in[16];
  const float* W2      = (const float*)d_in[17];
  const float* b2v     = (const float*)d_in[18];
  const float* Wmu     = (const float*)d_in[19];
  const float* bmu     = (const float*)d_in[20];
  const float* Wsig    = (const float*)d_in[21];
  const float* bsig    = (const float*)d_in[22];
  const float* Wu      = (const float*)d_in[23];
  const float* bu      = (const float*)d_in[24];
  const float* Wc      = (const float*)d_in[25];
  const float* bc      = (const float*)d_in[26];
  float* out = (float*)d_out;

  float* ws = (float*)d_ws;
  size_t off = 0;
  float* sac    = ws + off; off += 128;
  float* somac  = ws + off; off += 128;
  float* h      = ws + off; off += (size_t)B_ * DM * NN;
  float* z      = ws + off; off += (size_t)B_ * ZC * NN;
  float* upre   = ws + off; off += (size_t)B_ * DM * NN;
  float* cpre   = ws + off; off += (size_t)B_ * DM * NN;
  float* inp    = ws + off; off += (size_t)NB * DM;
  float* y1     = ws + off; off += (size_t)NB * DM;
  float* tmp    = ws + off; off += (size_t)NB * DM;
  float* tmp2   = ws + off; off += (size_t)NB * DM;
  float* mu     = ws + off; off += (size_t)NB * DM;
  float* sig    = ws + off; off += (size_t)NB * DM;
  float* qkv    = ws + off; off += (size_t)NB * QKVD;
  float* scores = ws + off; off += (size_t)(B_ * NHEADS) * SC * NN;

  k_coeffs<<<1, 32, 0, stream>>>(sac, somac);
  { int n = B_ * DM * NN; k_zero<<<(n + 255) / 256, 256, 0, stream>>>(h, n); }

  const float isq = 1.0f / sqrtf((float)DH);
  const long long zb   = (long long)ZC * NN;     // per-b stride of z
  const long long slot = (long long)CIN * NN;    // one diffusion slot

  for (int t = 0; t < TT; ++t) {
    if (t > 0) {
      // ---- noise branch: transformer encoder over nodes + DDPM-style update ----
      int nel = NB * DM;
      k_build_inp<<<(nel + 255) / 256, 256, 0, stream>>>(h, emb, mask, inp, t);
      // qkv = inp @ Wqkv^T + bqkv
      launch_gemm(stream, inp, Wqkv, qkv, bqkv, NB, QKVD, DM, DM, DM, QKVD, 1,
                  1, 1, 0, 0, 0, 0, 0, 0, 2, 0, 1.f);
      // attention in 4 s-chunks through one scores buffer, batched over (b,h)
      for (int ci = 0; ci < NCHUNK; ++ci) {
        const float* qb = qkv + (size_t)ci * SC * (B_ * QKVD);
        launch_gemm(stream, qb, qkv + DM, scores, nullptr,
                    SC, NN, DH, B_ * QKVD, B_ * QKVD, NN, 1,
                    B_ * NHEADS, NHEADS,
                    QKVD, DH, QKVD, DH,
                    (long long)NHEADS * SC * NN, (long long)SC * NN, 0, 0, isq);
        int rows = B_ * NHEADS * SC;
        k_softmax<<<(rows + 3) / 4, 128, 0, stream>>>(scores, rows, NN);
        launch_gemm(stream, scores, qkv + 2 * DM, tmp + (size_t)ci * SC * (B_ * DM), nullptr,
                    SC, DH, NN, NN, B_ * QKVD, B_ * DM, 0,
                    B_ * NHEADS, NHEADS,
                    (long long)NHEADS * SC * NN, (long long)SC * NN,
                    QKVD, DH, DM, DH, 0, 0, 1.f);
      }
      // Wo projection, post-norm LN1
      launch_gemm(stream, tmp, Wo, tmp2, bo, NB, DM, DM, DM, DM, DM, 1,
                  1, 1, 0, 0, 0, 0, 0, 0, 2, 0, 1.f);
      k_add_ln<<<(NB + 3) / 4, 128, 0, stream>>>(inp, tmp2, ln1g, ln1b, y1, NB);
      // FFN (relu fused), LN2 -> reuse inp as y
      launch_gemm(stream, y1, W1, tmp, b1v, NB, DM, DM, DM, DM, DM, 1,
                  1, 1, 0, 0, 0, 0, 0, 0, 2, 1, 1.f);
      launch_gemm(stream, tmp, W2, tmp2, b2v, NB, DM, DM, DM, DM, DM, 1,
                  1, 1, 0, 0, 0, 0, 0, 0, 2, 0, 1.f);
      k_add_ln<<<(NB + 3) / 4, 128, 0, stream>>>(y1, tmp2, ln2g, ln2b, inp, NB);
      // mu / sigma heads
      launch_gemm(stream, inp, Wmu, mu, bmu, NB, DM, DM, DM, DM, DM, 1,
                  1, 1, 0, 0, 0, 0, 0, 0, 2, 0, 1.f);
      launch_gemm(stream, inp, Wsig, sig, bsig, NB, DM, DM, DM, DM, DM, 1,
                  1, 1, 0, 0, 0, 0, 0, 0, 2, 0, 1.f);
      int n2 = B_ * DM * NN;
      k_noise<<<(n2 + 255) / 256, 256, 0, stream>>>(h, mu, sig, delta, sac, somac, t);
    }
    // ---- diffusion-conv GRU cell ----
    {
      int nel = B_ * CIN * NN;
      k_build_xg<<<(nel + 255) / 256, 256, 0, stream>>>(x, h, z, t);
      // G1 = G0 @ A0^T ; G2 = G1 @ A0^T ; G3 = G0 @ A1^T ; G4 = G3 @ A1^T
      launch_gemm(stream, z,            adj,           z + slot,     nullptr, CIN, NN, NN, NN, NN, NN, 1,
                  B_, B_, 0, zb, 0, 0, 0, zb, 0, 0, 1.f);
      launch_gemm(stream, z + slot,     adj,           z + 2 * slot, nullptr, CIN, NN, NN, NN, NN, NN, 1,
                  B_, B_, 0, zb, 0, 0, 0, zb, 0, 0, 1.f);
      launch_gemm(stream, z,            adj + NN * NN, z + 3 * slot, nullptr, CIN, NN, NN, NN, NN, NN, 1,
                  B_, B_, 0, zb, 0, 0, 0, zb, 0, 0, 1.f);
      launch_gemm(stream, z + 3 * slot, adj + NN * NN, z + 4 * slot, nullptr, CIN, NN, NN, NN, NN, NN, 1,
                  B_, B_, 0, zb, 0, 0, 0, zb, 0, 0, 1.f);
      // u = sigmoid(Wu @ Z_b + bu), c = tanh(Wc @ Z_b + bc)  (activations fused)
      launch_gemm(stream, Wu, z, upre, bu, DM, NN, ZC, ZC, NN, NN, 0,
                  B_, B_, 0, 0, 0, zb, 0, (long long)DM * NN, 1, 2, 1.f);
      launch_gemm(stream, Wc, z, cpre, bc, DM, NN, ZC, ZC, NN, NN, 0,
                  B_, B_, 0, 0, 0, zb, 0, (long long)DM * NN, 1, 3, 1.f);
      int n2 = B_ * DM * NN;
      k_combine<<<(n2 + 255) / 256, 256, 0, stream>>>(upre, cpre, h, out, lengths, t);
    }
  }
  (void)in_sizes; (void)n_in; (void)out_size; (void)ws_size;
}
